// SemanticLSTM_85186381349210
// MI455X (gfx1250) — matrix operand order repaired and checked by reading, not compiled, not run on hardware
//
#include <hip/hip_runtime.h>
#include <stdint.h>

// ---------------- problem constants ----------------
#define B_    64
#define T_    80
#define HID   512
#define VOCAB 10000
#define SEMD  300
#define CNND  2048
#define TSTEPS 79   // T-1

// ---------------- WMMA types ----------------
typedef __attribute__((ext_vector_type(16))) __bf16 vbf16;
typedef __attribute__((ext_vector_type(8)))  float  vf8;

union ABFrag { vbf16 v; uint32_t u[8]; };

// Branchless float -> bf16 (round to nearest even, quiet NaN) — avoids EXEC-mask
// divergence in the epilogues.
__device__ __forceinline__ unsigned short f2bf(float f) {
  uint32_t u = __float_as_uint(f);
  uint32_t r = u + 0x7fffu + ((u >> 16) & 1u);
  bool isnan = (u & 0x7fffffffu) > 0x7f800000u;
  r = isnan ? (u | 0x00400000u) : r;
  return (unsigned short)(r >> 16);
}

// A fragment: 16x32 bf16 tile of a row-major (stride 512) bf16 activation.
// lane l<16 holds row m0+l, K = k0+{0..7} and k0+{16..23};
// lane l+16 holds same row, K = k0+{8..15} and k0+{24..31}.
// Both 4-dword groups are 16B aligned -> two global_load_b128.
__device__ __forceinline__ vbf16 load_a_frag(const uint32_t* rowp, int k0, int hf) {
  ABFrag t;
  int base = (k0 >> 1) + (hf << 2);
#pragma unroll
  for (int v = 0; v < 4; ++v) t.u[v] = rowp[base + v];
#pragma unroll
  for (int v = 0; v < 4; ++v) t.u[4 + v] = rowp[base + 8 + v];
  return t.v;
}

// B fragment from packed layout [ntile][kblock][lane][8 dwords]:
// each lane reads 8 *contiguous* dwords (32B, 32B-aligned) -> two global_load_b128,
// the whole wave reads one contiguous 1KB block.
__device__ __forceinline__ vbf16 load_b_frag(const uint32_t* blockBase, int lane) {
  ABFrag t;
  const uint32_t* p = blockBase + lane * 8;
#pragma unroll
  for (int v = 0; v < 8; ++v) t.u[v] = p[v];
  return t.v;
}

__device__ __forceinline__ vf8 wmma_bf16(vbf16 a, vbf16 b, vf8 c) {
  return __builtin_amdgcn_wmma_f32_16x16x32_bf16(false, a, false, b, (short)0, c, false, false);
}

__device__ __forceinline__ float sigm(float x) { return 1.0f / (1.0f + __expf(-x)); }

// ---------------- pack / convert kernels ----------------

// W: (G,K,N) f32 row-major -> packed bf16-pair dwords, layout per gate:
//   [N/16 (ntile)][K/32 (kblock)][32 (lane)][8 (v)]
// where lane = hf*16 + l : column = ntile*16+l, K-pair index kk = kblock*16 + (v>>2)*8 + hf*4 + (v&3)
// (matches A fragment layout: lanes 0-15 hold K 0..7,16..23, lanes 16-31 hold K 8..15,24..31)
__global__ void slstm_pack_b(const float* __restrict__ W, uint32_t* __restrict__ out,
                             int G, int K, int N) {
  size_t idx = (size_t)blockIdx.x * blockDim.x + threadIdx.x;
  int KB = K >> 5;
  size_t total = (size_t)G * (N >> 4) * KB * 256;
  if (idx >= total) return;
  int v = (int)(idx & 7);
  size_t t1 = idx >> 3;
  int lane = (int)(t1 & 31); t1 >>= 5;
  int kb = (int)(t1 % KB); t1 /= KB;
  int nt = (int)(t1 % (N >> 4));
  int g  = (int)(t1 / (N >> 4));
  int l = lane & 15, hf = lane >> 4;
  int col = nt * 16 + l;
  int kk = kb * 16 + (v >> 2) * 8 + hf * 4 + (v & 3);  // K-pair index
  const float* Wg = W + (size_t)g * K * N;
  float lo = Wg[(size_t)(2 * kk) * N + col];
  float hi = Wg[(size_t)(2 * kk + 1) * N + col];
  out[idx] = (uint32_t)f2bf(lo) | ((uint32_t)f2bf(hi) << 16);
}

__global__ void slstm_conv_bf(const float* __restrict__ in, unsigned short* __restrict__ out,
                              size_t n) {
  size_t i = (size_t)blockIdx.x * blockDim.x + threadIdx.x;
  if (i < n) out[i] = f2bf(in[i]);
}

// ---------------- one-time precompute (f32) ----------------

// out[g][b][i] = sum_s W[g,s,i] * x[b,s]     out laid out as idx = g*32768 + b*512 + i
__global__ void slstm_proj(const float* __restrict__ W, const float* __restrict__ x,
                           float* __restrict__ out, int K) {
  int idx = blockIdx.x * blockDim.x + threadIdx.x;
  if (idx >= 4 * B_ * HID) return;
  int i = idx & 511;
  int b = (idx >> 9) & 63;
  int g = idx >> 15;
  const float* w  = W + (size_t)g * K * HID + i;
  const float* xb = x + (size_t)b * K;
  float s = 0.0f;
  for (int k = 0; k < K; ++k) s = fmaf(xb[k], w[(size_t)k * HID], s);
  out[idx] = s;
}

// vfeat[g][b][h] = sum_i (av*bv)[g][b][i] * Cc[g,i,h]
__global__ void slstm_vfeat(const float* __restrict__ av, const float* __restrict__ bv,
                            const float* __restrict__ Cc, float* __restrict__ vf) {
  int idx = blockIdx.x * blockDim.x + threadIdx.x;
  if (idx >= 4 * B_ * HID) return;
  int h = idx & 511;
  int base = idx & ~511;          // (g*64+b)*512
  int g = idx >> 15;
  const float* c = Cc + (size_t)g * HID * HID + h;
  float s = 0.0f;
  for (int i = 0; i < HID; ++i)
    s = fmaf(av[base + i] * bv[base + i], c[(size_t)i * HID], s);
  vf[idx] = s;
}

__global__ void slstm_init_state(float* __restrict__ ct, unsigned short* __restrict__ htbf) {
  int idx = blockIdx.x * blockDim.x + threadIdx.x;
  if (idx >= B_ * HID) return;
  ct[idx] = 0.0f;
  htbf[idx] = 0;
}

__global__ void slstm_init_cap(const int* __restrict__ captions, int* __restrict__ cap) {
  int b = threadIdx.x;
  if (b < B_) cap[b] = captions[b * T_];
}

// ---------------- per-step kernels ----------------

// S1: u[g][b][i] = (xemb[b]@Wa[g]) * bx[g][b][i]  ;  w[g][b][i] = (ht[b]@Ua[g]) * bh[g][b][i]
// 256 waves: which(2) x g(4) x mTile(4) x n64Tile(8)
__global__ void __launch_bounds__(256)
slstm_s1(const int* __restrict__ cap,
         const unsigned short* __restrict__ embed_bf,
         const unsigned short* __restrict__ ht_bf,
         const uint32_t* __restrict__ packWa,
         const uint32_t* __restrict__ packUa,
         const float* __restrict__ bx, const float* __restrict__ bh,
         unsigned short* __restrict__ u_bf, unsigned short* __restrict__ w_bf) {
  int wave = blockIdx.x * (blockDim.x >> 5) + (threadIdx.x >> 5);
  int lane = threadIdx.x & 31;
  int l = lane & 15, hf = lane >> 4;
  int which = wave >> 7;
  int rem = wave & 127;
  int g  = rem >> 5;
  int m0 = ((rem >> 3) & 3) * 16;
  int n0 = (rem & 7) * 64;

  int row_b = m0 + l;
  const uint32_t* rowp;
  if (which == 0) {
    int c = cap[row_b];
    rowp = (const uint32_t*)(embed_bf + (size_t)c * HID);
  } else {
    rowp = (const uint32_t*)(ht_bf + (size_t)row_b * HID);
  }
  const uint32_t* packW = (which == 0 ? packWa : packUa) + (size_t)g * (256 * 512);

  vf8 acc[4];
#pragma unroll
  for (int j = 0; j < 4; ++j) acc[j] = (vf8){};

#pragma unroll 4
  for (int k0 = 0; k0 < HID; k0 += 32) {
    vbf16 a = load_a_frag(rowp, k0, hf);
    int kb = k0 >> 5;
#pragma unroll
    for (int j = 0; j < 4; ++j) {
      vbf16 b = load_b_frag(packW + (size_t)((n0 >> 4) + j) * 4096 + kb * 256, lane);
      acc[j] = wmma_bf16(a, b, acc[j]);
    }
  }

  const float* scale = (which == 0 ? bx : bh) + (size_t)g * B_ * HID;
  unsigned short* outp = (which == 0 ? u_bf : w_bf) + (size_t)g * B_ * HID;
#pragma unroll
  for (int j = 0; j < 4; ++j) {
#pragma unroll
    for (int r = 0; r < 8; ++r) {
      int row = m0 + hf * 8 + r;
      int col = n0 + j * 16 + l;
      float v = acc[j][r] * scale[row * HID + col];
      outp[row * HID + col] = f2bf(v);
    }
  }
}

// S2: pre[g][b][h] = u[g][b]@Wc[g] + w[g][b]@Uc[g] + vfeat + bias   (128 waves)
__global__ void __launch_bounds__(256)
slstm_s2(const unsigned short* __restrict__ u_bf,
         const unsigned short* __restrict__ w_bf,
         const uint32_t* __restrict__ packWc,
         const uint32_t* __restrict__ packUc,
         const float* __restrict__ vfeat, const float* __restrict__ bias,
         float* __restrict__ pre) {
  int wave = blockIdx.x * (blockDim.x >> 5) + (threadIdx.x >> 5);
  int lane = threadIdx.x & 31;
  int l = lane & 15, hf = lane >> 4;
  int g  = wave >> 5;
  int m0 = ((wave >> 3) & 3) * 16;
  int n0 = (wave & 7) * 64;

  const uint32_t* rowU = (const uint32_t*)(u_bf + ((size_t)g * B_ + m0 + l) * HID);
  const uint32_t* rowW = (const uint32_t*)(w_bf + ((size_t)g * B_ + m0 + l) * HID);
  const uint32_t* pWc = packWc + (size_t)g * (256 * 512);
  const uint32_t* pUc = packUc + (size_t)g * (256 * 512);

  vf8 acc[4];
#pragma unroll
  for (int j = 0; j < 4; ++j) acc[j] = (vf8){};

#pragma unroll 4
  for (int k0 = 0; k0 < HID; k0 += 32) {
    vbf16 a = load_a_frag(rowU, k0, hf);
    int kb = k0 >> 5;
#pragma unroll
    for (int j = 0; j < 4; ++j) {
      vbf16 b = load_b_frag(pWc + (size_t)((n0 >> 4) + j) * 4096 + kb * 256, lane);
      acc[j] = wmma_bf16(a, b, acc[j]);
    }
  }
#pragma unroll 4
  for (int k0 = 0; k0 < HID; k0 += 32) {
    vbf16 a = load_a_frag(rowW, k0, hf);
    int kb = k0 >> 5;
#pragma unroll
    for (int j = 0; j < 4; ++j) {
      vbf16 b = load_b_frag(pUc + (size_t)((n0 >> 4) + j) * 4096 + kb * 256, lane);
      acc[j] = wmma_bf16(a, b, acc[j]);
    }
  }

#pragma unroll
  for (int j = 0; j < 4; ++j) {
#pragma unroll
    for (int r = 0; r < 8; ++r) {
      int row = m0 + hf * 8 + r;
      int col = n0 + j * 16 + l;
      size_t o = ((size_t)g * B_ + row) * HID + col;
      pre[o] = acc[j][r] + vfeat[o] + bias[g * HID + col];
    }
  }
}

// S3: pointwise LSTM gates, ct f32, ht -> bf16
__global__ void slstm_s3(const float* __restrict__ pre, float* __restrict__ ct,
                         unsigned short* __restrict__ ht_bf) {
  int idx = blockIdx.x * blockDim.x + threadIdx.x;
  if (idx >= B_ * HID) return;
  float ig = sigm(pre[idx]);
  float fg = sigm(pre[32768 + idx]);
  float og = sigm(pre[65536 + idx]);
  float gg = tanhf(pre[98304 + idx]);
  float c = fg * ct[idx] + ig * gg;
  ct[idx] = c;
  float h = og * tanhf(c);
  ht_bf[idx] = f2bf(h);
}

// S4: logits = ht @ W_last + b_last  -> d_out[:, t, :]     (2500 waves, 16x16 tiles)
__global__ void __launch_bounds__(256)
slstm_s4(const unsigned short* __restrict__ ht_bf,
         const uint32_t* __restrict__ packWL,
         const float* __restrict__ b_last,
         float* __restrict__ out, int t) {
  int wave = blockIdx.x * (blockDim.x >> 5) + (threadIdx.x >> 5);
  int mi = wave & 3;
  int nt = wave >> 2;
  if (nt >= VOCAB / 16) return;
  int lane = threadIdx.x & 31;
  int l = lane & 15, hf = lane >> 4;
  int m0 = mi * 16;
  int n0 = nt * 16;

  const uint32_t* rowp = (const uint32_t*)(ht_bf + (size_t)(m0 + l) * HID);
  const uint32_t* tile = packWL + (size_t)nt * 4096;

  vf8 acc = (vf8){};
#pragma unroll 4
  for (int k0 = 0; k0 < HID; k0 += 32) {
    vbf16 a = load_a_frag(rowp, k0, hf);
    vbf16 b = load_b_frag(tile + (k0 >> 5) * 256, lane);
    acc = wmma_bf16(a, b, acc);
  }

  float bl = b_last[n0 + l];
#pragma unroll
  for (int r = 0; r < 8; ++r) {
    int b = m0 + hf * 8 + r;
    out[((size_t)b * TSTEPS + t) * VOCAB + n0 + l] = acc[r] + bl;
  }
}

// S5: first-index argmax over 10000 logits per batch row -> cap
__global__ void slstm_s5(const float* __restrict__ out, int* __restrict__ cap, int t) {
  __shared__ float sv[256];
  __shared__ int   si[256];
  int b = blockIdx.x;
  int tid = threadIdx.x;
  const float* row = out + ((size_t)b * TSTEPS + t) * VOCAB;
  float best = -3.402823466e+38f;
  int bi = VOCAB;
  for (int n = tid; n < VOCAB; n += 256) {
    float v = row[n];
    if (v > best) { best = v; bi = n; }
  }
  sv[tid] = best; si[tid] = bi;
  __syncthreads();
  for (int s = 128; s > 0; s >>= 1) {
    if (tid < s) {
      if (sv[tid + s] > sv[tid] || (sv[tid + s] == sv[tid] && si[tid + s] < si[tid])) {
        sv[tid] = sv[tid + s]; si[tid] = si[tid + s];
      }
    }
    __syncthreads();
  }
  if (tid == 0) cap[b] = si[0];
}

// ---------------- host launcher ----------------
extern "C" void kernel_launch(void* const* d_in, const int* in_sizes, int n_in,
                              void* d_out, int out_size, void* d_ws, size_t ws_size,
                              hipStream_t stream) {
  (void)in_sizes; (void)n_in; (void)out_size; (void)ws_size;

  const int*   captions = (const int*)  d_in[0];
  const float* cnn      = (const float*)d_in[1];
  const float* sem      = (const float*)d_in[2];
  const float* Wa   = (const float*)d_in[3];
  const float* Wb   = (const float*)d_in[4];
  const float* Wc   = (const float*)d_in[5];
  const float* Ca   = (const float*)d_in[6];
  const float* Cb   = (const float*)d_in[7];
  const float* Cc   = (const float*)d_in[8];
  const float* Ua   = (const float*)d_in[9];
  const float* Ub   = (const float*)d_in[10];
  const float* Uc   = (const float*)d_in[11];
  const float* bias = (const float*)d_in[12];
  const float* embed  = (const float*)d_in[13];
  const float* W_last = (const float*)d_in[14];
  const float* b_last = (const float*)d_in[15];
  float* out = (float*)d_out;

  char* ws = (char*)d_ws;
  size_t off = 0;
  auto take = [&](size_t bytes) -> char* {
    char* p = ws + off;
    off = (off + bytes + 255) & ~(size_t)255;
    return p;
  };

  const size_t GATE_PACK_BYTES = (size_t)4 * 256 * 512 * 4;   // 2 MB each
  uint32_t* packWa = (uint32_t*)take(GATE_PACK_BYTES);
  uint32_t* packWc = (uint32_t*)take(GATE_PACK_BYTES);
  uint32_t* packUa = (uint32_t*)take(GATE_PACK_BYTES);
  uint32_t* packUc = (uint32_t*)take(GATE_PACK_BYTES);
  uint32_t* packWL = (uint32_t*)take((size_t)256 * VOCAB * 4);          // 10.24 MB
  unsigned short* embed_bf = (unsigned short*)take((size_t)VOCAB * HID * 2); // 10.24 MB
  float* bx    = (float*)take((size_t)4 * B_ * HID * 4);
  float* bh    = (float*)take((size_t)4 * B_ * HID * 4);
  float* bv    = (float*)take((size_t)4 * B_ * HID * 4);
  float* av    = (float*)take((size_t)4 * B_ * HID * 4);
  float* vfeat = (float*)take((size_t)4 * B_ * HID * 4);
  unsigned short* u_bf = (unsigned short*)take((size_t)4 * B_ * HID * 2);
  unsigned short* w_bf = (unsigned short*)take((size_t)4 * B_ * HID * 2);
  float* pre = (float*)take((size_t)4 * B_ * HID * 4);
  unsigned short* ht_bf = (unsigned short*)take((size_t)B_ * HID * 2);
  float* ct  = (float*)take((size_t)B_ * HID * 4);
  int* cap   = (int*)take(256);

  // ---- per-call weight packing (bf16) ----
  slstm_pack_b<<<2048, 256, 0, stream>>>(Wa, packWa, 4, 512, 512);
  slstm_pack_b<<<2048, 256, 0, stream>>>(Wc, packWc, 4, 512, 512);
  slstm_pack_b<<<2048, 256, 0, stream>>>(Ua, packUa, 4, 512, 512);
  slstm_pack_b<<<2048, 256, 0, stream>>>(Uc, packUc, 4, 512, 512);
  slstm_pack_b<<<10000, 256, 0, stream>>>(W_last, packWL, 1, 512, VOCAB);
  slstm_conv_bf<<<20000, 256, 0, stream>>>(embed, embed_bf, (size_t)VOCAB * HID);

  // ---- one-time precompute ----
  slstm_proj<<<512, 256, 0, stream>>>(Wb, sem, bx, SEMD);
  slstm_proj<<<512, 256, 0, stream>>>(Ub, sem, bh, SEMD);
  slstm_proj<<<512, 256, 0, stream>>>(Cb, sem, bv, SEMD);
  slstm_proj<<<512, 256, 0, stream>>>(Ca, cnn, av, CNND);
  slstm_vfeat<<<512, 256, 0, stream>>>(av, bv, Cc, vfeat);
  slstm_init_state<<<128, 256, 0, stream>>>(ct, ht_bf);
  slstm_init_cap<<<1, 64, 0, stream>>>(captions, cap);

  // ---- sequential timesteps ----
  for (int t = 0; t < TSTEPS; ++t) {
    slstm_s1<<<32, 256, 0, stream>>>(cap, embed_bf, ht_bf, packWa, packUa, bx, bh, u_bf, w_bf);
    slstm_s2<<<16, 256, 0, stream>>>(u_bf, w_bf, packWc, packUc, vfeat, bias, pre);
    slstm_s3<<<128, 256, 0, stream>>>(pre, ct, ht_bf);
    slstm_s4<<<313, 256, 0, stream>>>(ht_bf, packWL, b_last, out, t);
    slstm_s5<<<64, 256, 0, stream>>>(out, cap, t);
  }
}
